// TbDNet_55903294325290
// MI455X (gfx1250) — compile-verified, matrix operand order checked by reading, not run on hardware
//
#include <hip/hip_runtime.h>

// Problem constants (match reference)
#define B_    64
#define S_    32
#define N_    128
#define DV_   512
#define DW_   300
#define DWP_  320     // DW padded to multiple of 32 for WMMA K
#define DH_   1024
#define CLS_  1024
#define TCTRL_ 12
#define NE_   16
#define NA_   28

typedef __attribute__((ext_vector_type(16))) _Float16 v16h;
typedef __attribute__((ext_vector_type(8)))  _Float16 v8h;
typedef __attribute__((ext_vector_type(8)))  float    v8f;
typedef __attribute__((ext_vector_type(4)))  float    v4f;

__device__ __forceinline__ float sigmoidf_(float x) { return 1.0f / (1.0f + __expf(-x)); }

// ---- WMMA helpers (CDNA5 v_wmma_f32_16x16x32_f16, wave32) ----
__device__ __forceinline__ v8f wmma16(v16h a, v16h b, v8f c) {
  return __builtin_amdgcn_wmma_f32_16x16x32_f16(false, a, false, b, (short)0, c, false, false);
}

// A fragment from f32 row-major A (M x K). Row = row0 + (lane&15).
// K halves: base..base+7 and base+16..base+23, base = (lane&16)?8:0.
__device__ __forceinline__ v16h load_a_f32(const float* __restrict__ A, int lda,
                                           int row0, int k0, int lane) {
  const float* p = A + (size_t)(row0 + (lane & 15)) * lda + k0 + ((lane & 16) ? 8 : 0);
  v4f x0 = *(const v4f*)(p);
  v4f x1 = *(const v4f*)(p + 4);
  v4f y0 = *(const v4f*)(p + 16);
  v4f y1 = *(const v4f*)(p + 20);
  v16h a;
#pragma unroll
  for (int e = 0; e < 4; ++e) {
    a[e]      = (_Float16)x0[e];
    a[4 + e]  = (_Float16)x1[e];
    a[8 + e]  = (_Float16)y0[e];
    a[12 + e] = (_Float16)y1[e];
  }
  return a;
}

// Fragment from f16 row-major matrix (row-index matrix x K): used for both the
// pre-transposed weights (B side) and f16 activations (A side). Two b128 loads.
__device__ __forceinline__ v16h load_f16frag(const _Float16* __restrict__ Bt, int ldb,
                                             int row0, int k0, int lane) {
  const _Float16* p = Bt + (size_t)(row0 + (lane & 15)) * ldb + k0 + ((lane & 16) ? 8 : 0);
  v8h b0 = *(const v8h*)(p);
  v8h b1 = *(const v8h*)(p + 16);
  v16h b;
#pragma unroll
  for (int e = 0; e < 8; ++e) { b[e] = b0[e]; b[8 + e] = b1[e]; }
  return b;
}

// ---- Generic WMMA GEMM: C(MxN,f32) = act(A(MxK,f32) * Bt^T + bias) ----
// Each wave computes a 16x64 tile. Software-pipelined: next K-step's fragments
// are loaded (into distinct named registers) BEFORE the WMMAs that consume the
// previous K-step's fragments, so WMMA issue overlaps loads in flight and the
// scheduler can use partial s_wait_loadcnt.  Requires K >= 32 (always true here).
// ACT: 0=none, 1=relu, 2=tanh
template <int ACT>
__global__ __launch_bounds__(256) void gemm_wmma_k(
    const float* __restrict__ A, const _Float16* __restrict__ Bt,
    const float* __restrict__ bias, float* __restrict__ C,
    int M, int N, int K, int lda, int ldb, int ldc) {
  int gw = (int)((blockIdx.x * blockDim.x + threadIdx.x) >> 5);
  int lane = threadIdx.x & 31;
  int ntile = N >> 6;
  int total = (M >> 4) * ntile;
  if (gw >= total) return;
  int row0 = (gw / ntile) << 4;
  int col0 = (gw % ntile) << 6;
  v8f acc0, acc1, acc2, acc3;
#pragma unroll
  for (int r = 0; r < 8; ++r) { acc0[r] = 0.0f; acc1[r] = 0.0f; acc2[r] = 0.0f; acc3[r] = 0.0f; }
  // pipeline prologue: fragments for k0 = 0
  v16h a  = load_a_f32(A, lda, row0, 0, lane);
  v16h b0 = load_f16frag(Bt, ldb, col0,      0, lane);
  v16h b1 = load_f16frag(Bt, ldb, col0 + 16, 0, lane);
  v16h b2 = load_f16frag(Bt, ldb, col0 + 32, 0, lane);
  v16h b3 = load_f16frag(Bt, ldb, col0 + 48, 0, lane);
#pragma unroll 2
  for (int k0 = 32; k0 < K; k0 += 32) {
    v16h an = load_a_f32(A, lda, row0, k0, lane);
    v16h c0 = load_f16frag(Bt, ldb, col0,      k0, lane);
    v16h c1 = load_f16frag(Bt, ldb, col0 + 16, k0, lane);
    v16h c2 = load_f16frag(Bt, ldb, col0 + 32, k0, lane);
    v16h c3 = load_f16frag(Bt, ldb, col0 + 48, k0, lane);
    acc0 = wmma16(a, b0, acc0);
    acc1 = wmma16(a, b1, acc1);
    acc2 = wmma16(a, b2, acc2);
    acc3 = wmma16(a, b3, acc3);
    a = an; b0 = c0; b1 = c1; b2 = c2; b3 = c3;
  }
  // pipeline epilogue: last K-step
  acc0 = wmma16(a, b0, acc0);
  acc1 = wmma16(a, b1, acc1);
  acc2 = wmma16(a, b2, acc2);
  acc3 = wmma16(a, b3, acc3);
  int rbase = row0 + ((lane & 16) ? 8 : 0);
  int cl = lane & 15;
  v8f accs[4] = {acc0, acc1, acc2, acc3};
#pragma unroll
  for (int j = 0; j < 4; ++j) {
    int c = col0 + 16 * j + cl;
    float bv = bias ? bias[c] : 0.0f;
#pragma unroll
    for (int r = 0; r < 8; ++r) {
      float v = accs[j][r] + bv;
      if (ACT == 1) v = fmaxf(v, 0.0f);
      if (ACT == 2) v = tanhf(v);
      C[(size_t)(rbase + r) * ldc + c] = v;
    }
  }
}

// ---- LSTM step: h/c update fused with h@Wh WMMA (4 gate tiles per wave) ----
// A side (h_prev) is consumed as f16 (written by the previous step's epilogue).
// Software-pipelined like the GEMM: fragments for K-step k+32 are in flight
// while the WMMAs for K-step k execute. Output h is 64x1024 -> 256 waves.
__global__ __launch_bounds__(256) void lstm_step_k(
    const _Float16* __restrict__ hprev16, const _Float16* __restrict__ WhT,
    const float* __restrict__ xwx_t, const float* __restrict__ b_lstm,
    float* __restrict__ cstate, float* __restrict__ hout, _Float16* __restrict__ hout16) {
  int wave = (int)((blockIdx.x * blockDim.x + threadIdx.x) >> 5);
  int lane = threadIdx.x & 31;
  int row0 = (wave >> 6) << 4;   // batch tile
  int col0 = (wave & 63) << 4;   // hidden column tile (0..1008)
  v8f acc0, acc1, acc2, acc3;
#pragma unroll
  for (int r = 0; r < 8; ++r) { acc0[r] = 0.0f; acc1[r] = 0.0f; acc2[r] = 0.0f; acc3[r] = 0.0f; }
  v16h a  = load_f16frag(hprev16, DH_, row0, 0, lane);
  v16h b0 = load_f16frag(WhT, DH_, col0,            0, lane);
  v16h b1 = load_f16frag(WhT, DH_, col0 + DH_,      0, lane);
  v16h b2 = load_f16frag(WhT, DH_, col0 + 2 * DH_,  0, lane);
  v16h b3 = load_f16frag(WhT, DH_, col0 + 3 * DH_,  0, lane);
#pragma unroll 2
  for (int k0 = 32; k0 < DH_; k0 += 32) {
    v16h an = load_f16frag(hprev16, DH_, row0, k0, lane);
    v16h c0 = load_f16frag(WhT, DH_, col0,           k0, lane);
    v16h c1 = load_f16frag(WhT, DH_, col0 + DH_,     k0, lane);
    v16h c2 = load_f16frag(WhT, DH_, col0 + 2 * DH_, k0, lane);
    v16h c3 = load_f16frag(WhT, DH_, col0 + 3 * DH_, k0, lane);
    acc0 = wmma16(a, b0, acc0);
    acc1 = wmma16(a, b1, acc1);
    acc2 = wmma16(a, b2, acc2);
    acc3 = wmma16(a, b3, acc3);
    a = an; b0 = c0; b1 = c1; b2 = c2; b3 = c3;
  }
  acc0 = wmma16(a, b0, acc0);
  acc1 = wmma16(a, b1, acc1);
  acc2 = wmma16(a, b2, acc2);
  acc3 = wmma16(a, b3, acc3);
  int rbase = row0 + ((lane & 16) ? 8 : 0);
  int col = col0 + (lane & 15);
#pragma unroll
  for (int r = 0; r < 8; ++r) {
    int bb = rbase + r;
    const float* xr = xwx_t + (size_t)bb * (4 * DH_);
    float gi = acc0[r] + xr[col]            + b_lstm[col];
    float gf = acc1[r] + xr[col + DH_]      + b_lstm[col + DH_];
    float gg = acc2[r] + xr[col + 2 * DH_]  + b_lstm[col + 2 * DH_];
    float go = acc3[r] + xr[col + 3 * DH_]  + b_lstm[col + 3 * DH_];
    size_t idx = (size_t)bb * DH_ + col;
    float c = sigmoidf_(gf) * cstate[idx] + sigmoidf_(gi) * tanhf(gg);
    cstate[idx] = c;
    float h = sigmoidf_(go) * tanhf(c);
    hout[idx] = h;
    hout16[idx] = (_Float16)h;
  }
}

// ---- we = softmax(c_i @ E^T): single block, 4 waves, WMMA + half-wave softmax ----
__global__ __launch_bounds__(128) void we_k(const float* __restrict__ ci,
                                            const _Float16* __restrict__ EdgeH,
                                            float* __restrict__ we) {
  int wid = threadIdx.x >> 5, lane = threadIdx.x & 31;
  int row0 = wid * 16;
  v8f acc;
#pragma unroll
  for (int r = 0; r < 8; ++r) acc[r] = 0.0f;
  for (int k0 = 0; k0 < DV_; k0 += 32) {
    v16h a = load_a_f32(ci, DV_, row0, k0, lane);
    v16h b = load_f16frag(EdgeH, DV_, 0, k0, lane);
    acc = wmma16(a, b, acc);
  }
#pragma unroll
  for (int r = 0; r < 8; ++r) {
    float x = acc[r];
    float mx = x;
#pragma unroll
    for (int off = 1; off < 16; off <<= 1) mx = fmaxf(mx, __shfl_xor(mx, off, 32));
    float e = __expf(x - mx);
    float s = e;
#pragma unroll
    for (int off = 1; off < 16; off <<= 1) s += __shfl_xor(s, off, 32);
    int row = row0 + r + ((lane & 16) ? 8 : 0);
    we[row * NE_ + (lane & 15)] = e / s;
  }
}

// ---- Weight convert: out[n*Kp+k] = f16(in[k*N+n]) with zero K-padding ----
__global__ void convT_k(const float* __restrict__ in, _Float16* __restrict__ out,
                        int K, int N, int Kp) {
  size_t i = (size_t)blockIdx.x * blockDim.x + threadIdx.x;
  if (i >= (size_t)N * Kp) return;
  int n = (int)(i / Kp), k = (int)(i % Kp);
  out[i] = (k < K) ? (_Float16)in[(size_t)k * N + n] : (_Float16)0.0f;
}

__global__ void convF_k(const float* __restrict__ in, _Float16* __restrict__ out, int n) {
  int i = blockIdx.x * blockDim.x + threadIdx.x;
  if (i < n) out[i] = (_Float16)in[i];
}

__global__ void zero_k(float* p, int n) {
  int i = blockIdx.x * blockDim.x + threadIdx.x;
  if (i < n) p[i] = 0.0f;
}

__global__ void zero16_k(_Float16* p, int n) {
  int i = blockIdx.x * blockDim.x + threadIdx.x;
  if (i < n) p[i] = (_Float16)0.0f;
}

__global__ void init_stackptr_k(float* sp) {
  int i = blockIdx.x * blockDim.x + threadIdx.x;
  if (i < B_ * 4) sp[i] = ((i & 3) == 0) ? 1.0f : 0.0f;
}

// ---- emb gather (time-major, K-padded): embA[(s*B+b)*DWP + k] ----
__global__ void embgather_k(const int* __restrict__ questions,
                            const float* __restrict__ enc_emb,
                            float* __restrict__ embA) {
  size_t i = (size_t)blockIdx.x * blockDim.x + threadIdx.x;
  if (i >= (size_t)S_ * B_ * DWP_) return;
  int row = (int)(i / DWP_), k = (int)(i % DWP_);
  int s = row >> 6, b = row & 63;
  int q = questions[b * S_ + s];
  embA[i] = (k < DW_) ? enc_emb[(size_t)q * DW_ + k] : 0.0f;
}

// ---- feat gather: feat[(b*N+n)*DV + d] = attr_emb[v_idx[b*N+n]*DV + d] ----
__global__ void featgather_k(const int* __restrict__ v_idx,
                             const float* __restrict__ attr_emb,
                             float* __restrict__ feat) {
  size_t i = (size_t)blockIdx.x * blockDim.x + threadIdx.x;
  if (i >= (size_t)B_ * N_ * DV_) return;
  size_t row = i >> 9;
  int d = (int)(i & 511);
  feat[i] = attr_emb[(size_t)v_idx[row] * DV_ + d];
}

// ---- q_hidden gather into CQin[:,1024:2048] and Fin[:,512:1536] ----
__global__ void qhidden_k(const float* __restrict__ outs, const int* __restrict__ qlen,
                          float* __restrict__ CQin, float* __restrict__ Fin) {
  int b = blockIdx.x;
  int s = qlen[b] - 1;
  const float* src = outs + ((size_t)s * B_ + b) * DH_;
  for (int h = threadIdx.x; h < DH_; h += blockDim.x) {
    float v = src[h];
    CQin[(size_t)b * (2 * DH_) + DH_ + h] = v;
    Fin[(size_t)b * (DV_ + DH_) + DV_ + h] = v;
  }
}

// ---- module_prob = softmax(cq @ W_mod) ----
__global__ void modprob_k(const float* __restrict__ cq, const float* __restrict__ Wmod,
                          float* __restrict__ mprob) {
  int b = blockIdx.x;
  int wid = threadIdx.x >> 5, lane = threadIdx.x & 31;
  __shared__ float v[6];
  if (wid < 6) {
    float s = 0.0f;
    for (int k = lane; k < DH_; k += 32) s += cq[(size_t)b * DH_ + k] * Wmod[k * 6 + wid];
#pragma unroll
    for (int off = 16; off; off >>= 1) s += __shfl_down(s, off, 32);
    if (lane == 0) v[wid] = s;
  }
  __syncthreads();
  if (threadIdx.x == 0) {
    float mx = v[0];
    for (int m = 1; m < 6; ++m) mx = fmaxf(mx, v[m]);
    float e[6], sum = 0.0f;
    for (int m = 0; m < 6; ++m) { e[m] = __expf(v[m] - mx); sum += e[m]; }
    float inv = 1.0f / sum;
    for (int m = 0; m < 6; ++m) mprob[b * 6 + m] = e[m] * inv;
  }
}

// ---- attention: al/cv softmax over S + c_ctrl (written into CQin[:,0:1024]) ----
__global__ void attention_k(const float* __restrict__ outs, const float* __restrict__ catt,
                            const int* __restrict__ qlen, float* __restrict__ CQin) {
  int b = blockIdx.x;
  __shared__ float al[S_];
  __shared__ float cv[S_];
  int wid = threadIdx.x >> 5, lane = threadIdx.x & 31;
  const float* q = catt + (size_t)b * DH_;
  for (int s = wid; s < S_; s += 8) {
    const float* o = outs + ((size_t)s * B_ + b) * DH_;
    float sum = 0.0f;
    for (int k = lane; k < DH_; k += 32) sum += o[k] * q[k];
#pragma unroll
    for (int off = 16; off; off >>= 1) sum += __shfl_down(sum, off, 32);
    if (lane == 0) al[s] = sum;
  }
  __syncthreads();
  if (threadIdx.x == 0) {
    int L = qlen[b];
    float mx = -1e30f;
    for (int s = 0; s < L; ++s) mx = fmaxf(mx, al[s]);
    float sum = 0.0f;
    for (int s = 0; s < S_; ++s) {
      float e = (s < L) ? __expf(al[s] - mx) : 0.0f;
      cv[s] = e; sum += e;
    }
    float inv = 1.0f / sum;
    for (int s = 0; s < S_; ++s) cv[s] *= inv;
  }
  __syncthreads();
  for (int h = threadIdx.x; h < DH_; h += blockDim.x) {
    float acc = 0.0f;
    for (int s = 0; s < S_; ++s) acc += cv[s] * outs[((size_t)s * B_ + b) * DH_ + h];
    CQin[(size_t)b * (2 * DH_) + h] = acc;  // c_ctrl lives in CQin cols 0..1023
  }
}

// ---- a_find[b,n] = sigmoid(dot(featW[b,n,:], c_i[b,:]) / sqrt(DV)) ----
__global__ __launch_bounds__(256) void find_k(const float* __restrict__ featW,
                                              const float* __restrict__ ci,
                                              float* __restrict__ afind) {
  int w = (int)((blockIdx.x * blockDim.x + threadIdx.x) >> 5);
  int lane = threadIdx.x & 31;
  int b = w >> 7, n = w & 127;
  const float* f = featW + ((size_t)b * N_ + n) * DV_;
  const float* c = ci + (size_t)b * DV_;
  float s = 0.0f;
  for (int k = lane; k < DV_; k += 32) s += f[k] * c[k];
#pragma unroll
  for (int off = 16; off; off >>= 1) s += __shfl_down(s, off, 32);
  if (lane == 0) afind[b * N_ + n] = sigmoidf_(s * 0.044194173824159216f); // 1/sqrt(512)
}

// ---- a_in[b,n] = sum_l att_stack[b,n,l] * ptr[b,l] ----
__global__ void ain_k(const float* __restrict__ attstack, const float* __restrict__ sptr,
                      float* __restrict__ ain) {
  int i = blockIdx.x * blockDim.x + threadIdx.x;
  if (i >= B_ * N_) return;
  int b = i >> 7;
  float s = 0.0f;
#pragma unroll
  for (int l = 0; l < 4; ++l) s += attstack[(size_t)i * 4 + l] * sptr[b * 4 + l];
  ain[i] = s;
}

// ---- a_trans[b,i] = min(sum_j we[b,cat[b,i,j]]*conn[b,i,j]*a_in[b,j], 1) ----
__global__ __launch_bounds__(128) void atrans_k(const float* __restrict__ we,
                                                const int* __restrict__ cat_mat,
                                                const float* __restrict__ conn,
                                                const float* __restrict__ ain,
                                                float* __restrict__ atrans) {
  int b = blockIdx.x, i = threadIdx.x;
  const int* cm = cat_mat + ((size_t)b * N_ + i) * N_;
  const float* cn = conn + ((size_t)b * N_ + i) * N_;
  const float* ai = ain + b * N_;
  const float* w = we + b * NE_;
  float s = 0.0f;
  for (int j = 0; j < N_; ++j) s += w[cm[j]] * cn[j] * ai[j];
  atrans[b * N_ + i] = fminf(s, 1.0f);
}

// ---- pooled[b,d] = sum_n a_in*feat / (sum a_in + 1e-6) ----
__global__ __launch_bounds__(256) void pooled_k(const float* __restrict__ ain,
                                                const float* __restrict__ feat,
                                                float* __restrict__ pooled) {
  int b = blockIdx.x;
  __shared__ float av[N_];
  __shared__ float denom;
  if (threadIdx.x < N_) av[threadIdx.x] = ain[b * N_ + threadIdx.x];
  __syncthreads();
  if (threadIdx.x == 0) {
    float s = 0.0f;
    for (int n = 0; n < N_; ++n) s += av[n];
    denom = 1.0f / (s + 1e-6f);
  }
  __syncthreads();
  for (int d = threadIdx.x; d < DV_; d += blockDim.x) {
    float acc = 0.0f;
    for (int n = 0; n < N_; ++n) acc += av[n] * feat[((size_t)b * N_ + n) * DV_ + d];
    pooled[(size_t)b * DV_ + d] = acc * denom;
  }
}

// ---- stack / mem blend update (one block per batch) ----
__global__ __launch_bounds__(128) void update_k(const float* __restrict__ mprob,
                                                float* __restrict__ sptr,
                                                float* __restrict__ attstack,
                                                float* __restrict__ memb,
                                                const float* __restrict__ afind,
                                                const float* __restrict__ atrans,
                                                const float* __restrict__ ain,
                                                const float* __restrict__ ddesc) {
  int b = blockIdx.x, tid = threadIdx.x;
  __shared__ float mp[6], ptr[4], pfw[4], pbw[4];
  if (tid < 4) ptr[tid] = sptr[b * 4 + tid];
  __syncthreads();
  if (tid == 0) {
    float val[6];
    float p0 = ptr[0], p1 = ptr[1], p2 = ptr[2], p3 = ptr[3];
    val[0] = p0 + p1 + p2 + p3;
    val[1] = p0 + p1 + p2;
    val[2] = p1 + p2 + p3;
    val[3] = val[2];
    val[4] = p2 + p3;
    val[5] = val[2];
    float s = 0.0f;
    for (int m = 0; m < 6; ++m) { mp[m] = mprob[b * 6 + m] * roundf(val[m]); s += mp[m]; }
    float inv = (s > 0.0f) ? (1.0f / s) : 1.0f;
    for (int m = 0; m < 6; ++m) mp[m] *= inv;
    pfw[0] = 0.0f; pfw[1] = p0; pfw[2] = p1; pfw[3] = p2 + p3;
    pbw[0] = p0 + p1; pbw[1] = p2; pbw[2] = p3; pbw[3] = 0.0f;
  }
  __syncthreads();
  {
    int n = tid;  // 128 threads == N_
    float st[4];
#pragma unroll
    for (int l = 0; l < 4; ++l) st[l] = attstack[((size_t)b * N_ + n) * 4 + l];
    float a_in = ain[b * N_ + n];
    float a_f  = afind[b * N_ + n];
    float a_t  = atrans[b * N_ + n];
    float aprev = st[0] * pbw[0] + st[1] * pbw[1] + st[2] * pbw[2] + st[3] * pbw[3];
    float amin = fminf(aprev, a_in);
    float afin = a_in * a_f;
#pragma unroll
    for (int l = 0; l < 4; ++l) {
      float s0 = st[l];
      float s1 = a_f  * pfw[l] + s0 * (1.0f - pfw[l]);
      float s2 = a_t  * ptr[l] + s0 * (1.0f - ptr[l]);
      float s3 = afin * ptr[l] + s0 * (1.0f - ptr[l]);
      float s4 = amin * pbw[l] + s0 * (1.0f - pbw[l]);
      attstack[((size_t)b * N_ + n) * 4 + l] =
          mp[0] * s0 + mp[1] * s1 + mp[2] * s2 + mp[3] * s3 + mp[4] * s4 + mp[5] * s0;
    }
  }
  if (tid < 4) {
    int l = tid;
    sptr[b * 4 + l] = (mp[0] + mp[2] + mp[3]) * ptr[l] + mp[1] * pfw[l] +
                      (mp[4] + mp[5]) * pbw[l];
  }
  float keep = mp[0] + mp[1] + mp[2] + mp[3] + mp[4];
  for (int d = tid; d < DV_; d += 128) {
    size_t idx = (size_t)b * DV_ + d;
    memb[idx] = keep * memb[idx] + mp[5] * ddesc[idx];
  }
}

// ---- copy mem into Fin[:,0:512] ----
__global__ void finmem_k(const float* __restrict__ memb, float* __restrict__ Fin) {
  int i = blockIdx.x * blockDim.x + threadIdx.x;
  if (i >= B_ * DV_) return;
  int b = i >> 9, d = i & 511;
  Fin[(size_t)b * (DV_ + DH_) + d] = memb[i];
}

// ---- final logits: out = h1 @ W_cls2 + b_cls2 (64 x 28) ----
__global__ void cls2_k(const float* __restrict__ h1, const float* __restrict__ W2,
                       const float* __restrict__ b2, float* __restrict__ out) {
  int b = blockIdx.x, j = threadIdx.x;
  if (j >= NA_) return;
  float s = b2[j];
  for (int k = 0; k < CLS_; ++k) s += h1[(size_t)b * CLS_ + k] * W2[k * NA_ + j];
  out[b * NA_ + j] = s;
}

extern "C" void kernel_launch(void* const* d_in, const int* in_sizes, int n_in,
                              void* d_out, int out_size, void* d_ws, size_t ws_size,
                              hipStream_t stream) {
  (void)in_sizes; (void)n_in; (void)out_size; (void)ws_size;
  const int*   questions = (const int*)d_in[0];
  const int*   qlen      = (const int*)d_in[1];
  const float* conn      = (const float*)d_in[2];
  const int*   cat_mat   = (const int*)d_in[3];
  const int*   v_idx     = (const int*)d_in[4];
  const float* attr_emb  = (const float*)d_in[5];
  const float* edge_vec  = (const float*)d_in[6];
  const float* enc_emb   = (const float*)d_in[7];
  const float* Wx        = (const float*)d_in[8];
  const float* Wh        = (const float*)d_in[9];
  const float* b_lstm    = (const float*)d_in[10];
  const float* W_cq      = (const float*)d_in[11];
  const float* b_cq      = (const float*)d_in[12];
  const float* W_mod     = (const float*)d_in[13];
  const float* W_att     = (const float*)d_in[14];
  const float* W_q       = (const float*)d_in[15];
  const float* b_q       = (const float*)d_in[16];
  const float* W_find    = (const float*)d_in[17];
  const float* W_desc    = (const float*)d_in[18];
  const float* W_cls1    = (const float*)d_in[19];
  const float* b_cls1    = (const float*)d_in[20];
  const float* W_cls2    = (const float*)d_in[21];
  const float* b_cls2    = (const float*)d_in[22];
  float* out = (float*)d_out;

  // ---- workspace layout ----
  char* ws = (char*)d_ws;
  size_t off = 0;
  auto alloc = [&](size_t bytes) -> void* {
    void* p = ws + off;
    off = (off + bytes + 255) & ~(size_t)255;
    return p;
  };
  _Float16* WxT    = (_Float16*)alloc((size_t)4 * DH_ * DWP_ * 2);
  _Float16* WhT    = (_Float16*)alloc((size_t)4 * DH_ * DH_ * 2);
  _Float16* WcqT   = (_Float16*)alloc((size_t)DH_ * 2 * DH_ * 2);
  _Float16* WattT  = (_Float16*)alloc((size_t)DH_ * DH_ * 2);
  _Float16* WqT    = (_Float16*)alloc((size_t)DV_ * DH_ * 2);
  _Float16* WfindT = (_Float16*)alloc((size_t)DV_ * DV_ * 2);
  _Float16* WdescT = (_Float16*)alloc((size_t)DV_ * DV_ * 2);
  _Float16* Wcls1T = (_Float16*)alloc((size_t)CLS_ * (DV_ + DH_) * 2);
  _Float16* EdgeH  = (_Float16*)alloc((size_t)NE_ * DV_ * 2);
  _Float16* h16a   = (_Float16*)alloc((size_t)B_ * DH_ * 2);  // f16 h double buffer
  _Float16* h16b   = (_Float16*)alloc((size_t)B_ * DH_ * 2);
  float* embA   = (float*)alloc((size_t)S_ * B_ * DWP_ * 4);
  float* xWx    = (float*)alloc((size_t)S_ * B_ * 4 * DH_ * 4);
  float* outs   = (float*)alloc((size_t)S_ * B_ * DH_ * 4);
  float* cstate = (float*)alloc((size_t)B_ * DH_ * 4);
  float* CQin   = (float*)alloc((size_t)B_ * 2 * DH_ * 4);
  float* cq     = (float*)alloc((size_t)B_ * DH_ * 4);
  float* catt   = (float*)alloc((size_t)B_ * DH_ * 4);
  float* ci     = (float*)alloc((size_t)B_ * DV_ * 4);
  float* mprob  = (float*)alloc((size_t)B_ * 6 * 4);
  float* webuf  = (float*)alloc((size_t)B_ * NE_ * 4);
  float* feat   = (float*)alloc((size_t)B_ * N_ * DV_ * 4);
  float* featW  = (float*)alloc((size_t)B_ * N_ * DV_ * 4);
  float* afind  = (float*)alloc((size_t)B_ * N_ * 4);
  float* ain    = (float*)alloc((size_t)B_ * N_ * 4);
  float* atrans = (float*)alloc((size_t)B_ * N_ * 4);
  float* attstk = (float*)alloc((size_t)B_ * N_ * 4 * 4);
  float* sptr   = (float*)alloc((size_t)B_ * 4 * 4);
  float* memb   = (float*)alloc((size_t)B_ * DV_ * 4);
  float* pooled = (float*)alloc((size_t)B_ * DV_ * 4);
  float* ddesc  = (float*)alloc((size_t)B_ * DV_ * 4);
  float* Fin    = (float*)alloc((size_t)B_ * (DV_ + DH_) * 4);
  float* h1     = (float*)alloc((size_t)B_ * CLS_ * 4);

  auto eb = [](size_t n) { return (unsigned)((n + 255) / 256); };

  // ---- Phase 0: weight conversion (f32 -> transposed f16), gathers, zero-init ----
  convT_k<<<eb((size_t)4 * DH_ * DWP_), 256, 0, stream>>>(Wx, WxT, DW_, 4 * DH_, DWP_);
  convT_k<<<eb((size_t)4 * DH_ * DH_), 256, 0, stream>>>(Wh, WhT, DH_, 4 * DH_, DH_);
  convT_k<<<eb((size_t)DH_ * 2 * DH_), 256, 0, stream>>>(W_cq, WcqT, 2 * DH_, DH_, 2 * DH_);
  convT_k<<<eb((size_t)DH_ * DH_), 256, 0, stream>>>(W_att, WattT, DH_, DH_, DH_);
  convT_k<<<eb((size_t)DV_ * DH_), 256, 0, stream>>>(W_q, WqT, DH_, DV_, DH_);
  convT_k<<<eb((size_t)DV_ * DV_), 256, 0, stream>>>(W_find, WfindT, DV_, DV_, DV_);
  convT_k<<<eb((size_t)DV_ * DV_), 256, 0, stream>>>(W_desc, WdescT, DV_, DV_, DV_);
  convT_k<<<eb((size_t)CLS_ * (DV_ + DH_)), 256, 0, stream>>>(W_cls1, Wcls1T, DV_ + DH_, CLS_, DV_ + DH_);
  convF_k<<<eb((size_t)NE_ * DV_), 256, 0, stream>>>(edge_vec, EdgeH, NE_ * DV_);
  embgather_k<<<eb((size_t)S_ * B_ * DWP_), 256, 0, stream>>>(questions, enc_emb, embA);
  featgather_k<<<eb((size_t)B_ * N_ * DV_), 256, 0, stream>>>(v_idx, attr_emb, feat);
  zero16_k<<<eb((size_t)B_ * DH_), 256, 0, stream>>>(h16a, B_ * DH_);  // h_0 = 0
  zero_k<<<eb((size_t)B_ * DH_), 256, 0, stream>>>(cstate, B_ * DH_);
  zero_k<<<eb((size_t)B_ * 2 * DH_), 256, 0, stream>>>(CQin, B_ * 2 * DH_);
  zero_k<<<eb((size_t)B_ * N_ * 4), 256, 0, stream>>>(attstk, B_ * N_ * 4);
  zero_k<<<eb((size_t)B_ * DV_), 256, 0, stream>>>(memb, B_ * DV_);
  init_stackptr_k<<<1, 256, 0, stream>>>(sptr);

  // ---- Phase 1: hoisted GEMMs ----
  // xWx = embA(2048x320) @ WxT^T -> 2048x4096 (hoists x@Wx out of the recurrence)
  gemm_wmma_k<0><<<(2048 / 16) * (4096 / 64) / 8, 256, 0, stream>>>(
      embA, WxT, nullptr, xWx, S_ * B_, 4 * DH_, DWP_, DWP_, DWP_, 4 * DH_);
  // featW = feat(8192x512) @ WfindT^T -> 8192x512  (loop-invariant in reference)
  gemm_wmma_k<0><<<(8192 / 16) * (512 / 64) / 8, 256, 0, stream>>>(
      feat, WfindT, nullptr, featW, B_ * N_, DV_, DV_, DV_, DV_, DV_);

  // ---- Phase 2: LSTM recurrence (sequential, one fused WMMA kernel per step) ----
  for (int t = 0; t < S_; ++t) {
    const _Float16* hp16 = (t & 1) ? h16b : h16a;
    _Float16* ho16 = (t & 1) ? h16a : h16b;
    lstm_step_k<<<32, 256, 0, stream>>>(hp16, WhT, xWx + (size_t)t * B_ * 4 * DH_,
                                        b_lstm, cstate, outs + (size_t)t * B_ * DH_, ho16);
  }
  qhidden_k<<<B_, 256, 0, stream>>>(outs, qlen, CQin, Fin);

  // ---- Phase 3: control loop ----
  for (int t = 0; t < TCTRL_; ++t) {
    // cq = tanh([c_ctrl, q_hidden] @ W_cq + b_cq)
    gemm_wmma_k<2><<<(B_ / 16) * (DH_ / 64) / 8, 256, 0, stream>>>(
        CQin, WcqT, b_cq, cq, B_, DH_, 2 * DH_, 2 * DH_, 2 * DH_, DH_);
    modprob_k<<<B_, 192, 0, stream>>>(cq, W_mod, mprob);
    // catt = cq @ W_att
    gemm_wmma_k<0><<<(B_ / 16) * (DH_ / 64) / 8, 256, 0, stream>>>(
        cq, WattT, nullptr, catt, B_, DH_, DH_, DH_, DH_, DH_);
    attention_k<<<B_, 256, 0, stream>>>(outs, catt, qlen, CQin);  // new c_ctrl
    // c_i = relu(c_ctrl @ W_q + b_q)   (A = CQin cols 0..1023, lda=2048)
    gemm_wmma_k<1><<<(B_ / 16) * (DV_ / 64) / 8, 256, 0, stream>>>(
        CQin, WqT, b_q, ci, B_, DV_, DH_, 2 * DH_, DH_, DV_);
    we_k<<<1, 128, 0, stream>>>(ci, EdgeH, webuf);
    find_k<<<(B_ * N_) / 8, 256, 0, stream>>>(featW, ci, afind);
    ain_k<<<eb((size_t)B_ * N_), 256, 0, stream>>>(attstk, sptr, ain);
    atrans_k<<<B_, 128, 0, stream>>>(webuf, cat_mat, conn, ain, atrans);
    pooled_k<<<B_, 256, 0, stream>>>(ain, feat, pooled);
    // ddesc = pooled @ W_desc
    gemm_wmma_k<0><<<(B_ / 16) * (DV_ / 64) / 8, 256, 0, stream>>>(
        pooled, WdescT, nullptr, ddesc, B_, DV_, DV_, DV_, DV_, DV_);
    update_k<<<B_, 128, 0, stream>>>(mprob, sptr, attstk, memb, afind, atrans, ain, ddesc);
  }

  // ---- Phase 4: classifier ----
  finmem_k<<<eb((size_t)B_ * DV_), 256, 0, stream>>>(memb, Fin);
  gemm_wmma_k<1><<<(B_ / 16) * (CLS_ / 64) / 8, 256, 0, stream>>>(
      Fin, Wcls1T, b_cls1, h1, B_, CLS_, DV_ + DH_, DV_ + DH_, DV_ + DH_, CLS_);
  cls2_k<<<B_, 32, 0, stream>>>(h1, W_cls2, b_cls2, out);
}
